// MultiHeadSelfAttention_17360257810453
// MI455X (gfx1250) — matrix operand order repaired and checked by reading, not compiled, not run on hardware
//
#include <hip/hip_runtime.h>
#include <hip/hip_bf16.h>

#define S_LEN 1863
#define SP    1872            // S padded to multiple of 16 for Vt row stride (16B-aligned frag loads)
#define DM    1024
#define NH    16
#define HD    64
#define QT_COUNT 117          // ceil(S/16)
#define NEGF  (-1e30f)

typedef __bf16 bf16_t;
typedef __bf16 v16bf __attribute__((ext_vector_type(16)));
typedef float  v8f   __attribute__((ext_vector_type(8)));
typedef unsigned int u32x4 __attribute__((ext_vector_type(4)));

// round-to-nearest-even f32 -> bf16 (bit-exact, no reliance on __bf16 cvt legality)
__device__ __forceinline__ bf16_t f2bf(float f) {
  unsigned u = __float_as_uint(f);
  u += 0x7FFFu + ((u >> 16) & 1u);
  unsigned short h = (unsigned short)(u >> 16);
  bf16_t r;
  __builtin_memcpy(&r, &h, 2);
  return r;
}

// load a 16-element bf16 fragment as two 16-byte chunks
__device__ __forceinline__ v16bf load_frag2(const bf16_t* p0, const bf16_t* p1) {
  union { v16bf v; u32x4 q[2]; } f;
  f.q[0] = *(const u32x4*)p0;
  f.q[1] = *(const u32x4*)p1;
  return f.v;
}

// ---------------- f32 -> bf16 conversion ----------------
__global__ void k_f32_to_bf16(const float* __restrict__ src, bf16_t* __restrict__ dst, int n) {
  int i = blockIdx.x * blockDim.x + threadIdx.x;
  int stride = gridDim.x * blockDim.x;
  for (; i < n; i += stride) dst[i] = f2bf(src[i]);
}

// ---------------- WMMA GEMM: C[M,N] = A[M,K] * B[N,K]^T  (A,B bf16 row-major, C f32) ----
// block = 256 threads = 8 waves. Block tile: 256(M) x 64(N); wave tile 32(M) x 64(N).
// B tile (64x32 bf16 = 4KB) staged through LDS once per block per K-step.
__global__ void __launch_bounds__(256)
k_gemm_bf16(const bf16_t* __restrict__ A, const bf16_t* __restrict__ B,
            float* __restrict__ C, int M, int N, int K) {
  __shared__ __attribute__((aligned(16))) bf16_t Bs[64 * 32];
  const int lane = threadIdx.x & 31;
  const int wave = threadIdx.x >> 5;
  const int half = lane >> 4;
  const int l16  = lane & 15;
  const int m0 = blockIdx.x * 256 + wave * 32;
  const int n0 = blockIdx.y * 64;

  // clamped A row pointers (tail waves stay in the loop for the barriers; stores guarded)
  const bf16_t* Arow0 = A + (size_t)min(m0 + l16,      M - 1) * K;
  const bf16_t* Arow1 = A + (size_t)min(m0 + 16 + l16, M - 1) * K;

  // cooperative B-tile loader: thread t -> row t/4 (0..63), 16-byte chunk (t%4)*8
  const int tb_row = threadIdx.x >> 2;
  const int tb_col = (threadIdx.x & 3) * 8;
  const bf16_t* Bg = B + (size_t)(n0 + tb_row) * K + tb_col;
  bf16_t* BsW = &Bs[tb_row * 32 + tb_col];

  v8f acc[8] = {};
  for (int kb = 0; kb < K; kb += 32) {
    __syncthreads();                                   // previous step's frag reads done
    *(u32x4*)BsW = *(const u32x4*)(Bg + kb);           // stage 64x32 B tile
    if (kb + 32 < K) __builtin_prefetch(Bg + kb + 32, 0, 3);  // global_prefetch_b8 (near)
    __syncthreads();

    // A 16x32 frags: lane l16 = row, half selects K sub-stripes {h*8.., h*8+16..}
    v16bf a0 = load_frag2(Arow0 + kb + half * 8, Arow0 + kb + half * 8 + 16);
    v16bf a1 = load_frag2(Arow1 + kb + half * 8, Arow1 + kb + half * 8 + 16);

    // preload all four B 32x16 frags from LDS into distinct registers,
    // so the 8 ds_load_b128 issue as one clause and the 8 WMMAs run back-to-back
    v16bf bfrag[4];
#pragma unroll
    for (int j = 0; j < 4; ++j) {
      const bf16_t* Brow = &Bs[(j * 16 + l16) * 32 + half * 8];
      bfrag[j] = load_frag2(Brow, Brow + 16);
    }
#pragma unroll
    for (int j = 0; j < 4; ++j) {
      acc[j]     = __builtin_amdgcn_wmma_f32_16x16x32_bf16(false, a0, false, bfrag[j],
                                                           (short)0, acc[j], false, false);
      acc[j + 4] = __builtin_amdgcn_wmma_f32_16x16x32_bf16(false, a1, false, bfrag[j],
                                                           (short)0, acc[j + 4], false, false);
    }
  }
#pragma unroll
  for (int g = 0; g < 2; ++g)
#pragma unroll
    for (int j = 0; j < 4; ++j)
#pragma unroll
      for (int r = 0; r < 8; ++r) {
        int row = m0 + g * 16 + r + 8 * half;   // C layout: VGPR r holds M = r + 8*(lane/16)
        if (row < M) C[(size_t)row * N + n0 + j * 16 + l16] = acc[g * 4 + j][r];
      }
}

// ---------------- RoPE + pack (q scaled by 1/8, k bf16, V transposed per head) --------
__global__ void k_rope_pack(const float* __restrict__ qkv, const int* __restrict__ pos,
                            bf16_t* __restrict__ qb, bf16_t* __restrict__ kbuf,
                            bf16_t* __restrict__ vt) {
  int tid = blockIdx.x * blockDim.x + threadIdx.x;  // token*NH + head
  if (tid >= S_LEN * NH) return;
  const int s = tid / NH, h = tid % NH;
  const float* qp = qkv + (size_t)s * 3 * DM + h * HD;
  const float* kp = qp + DM;
  const float* vp = qp + 2 * DM;
  const int px = min(max(pos[s * 3 + 0], 0), 31);
  const int py = min(max(pos[s * 3 + 1], 0), 31);
  const int pz = min(max(pos[s * 3 + 2], 0), 7);

  bf16_t* qd = qb   + (size_t)s * DM + h * HD;
  bf16_t* kd = kbuf + (size_t)s * DM + h * HD;
#pragma unroll
  for (int p = 0; p < 32; ++p) {
    float fpos, expo;
    if (p < 10)      { fpos = (float)px; expo = (float)p        * (1.0f / 10.0f); }
    else if (p < 20) { fpos = (float)py; expo = (float)(p - 10) * (1.0f / 10.0f); }
    else             { fpos = (float)pz; expo = (float)(p - 20) * (1.0f / 12.0f); }
    float inv = __powf(10000.0f, -expo);
    float ang = fpos * inv;
    float c = __cosf(ang), sn = __sinf(ang);
    float q0 = qp[2 * p], q1 = qp[2 * p + 1];
    float k0 = kp[2 * p], k1 = kp[2 * p + 1];
    qd[2 * p]     = f2bf((q0 * c - q1 * sn) * 0.125f);   // fold HD^-0.5
    qd[2 * p + 1] = f2bf((q1 * c + q0 * sn) * 0.125f);
    kd[2 * p]     = f2bf(k0 * c - k1 * sn);
    kd[2 * p + 1] = f2bf(k1 * c + k0 * sn);
  }
#pragma unroll
  for (int d = 0; d < HD; ++d)
    vt[((size_t)h * HD + d) * SP + s] = f2bf(vp[d]);
}

// ---------------- flash attention: 1 wave per (head, 16-query tile) -------------------
__global__ void __launch_bounds__(256)
k_attention(const bf16_t* __restrict__ qb, const bf16_t* __restrict__ kbuf,
            const bf16_t* __restrict__ vt, const float* __restrict__ bias_x,
            const float* __restrict__ bias_y, const float* __restrict__ bias_z,
            const int* __restrict__ pos, bf16_t* __restrict__ attnb) {
  __shared__ __attribute__((aligned(16))) bf16_t plds[8][16 * 32]; // per-wave P tile
  const int lane = threadIdx.x & 31;
  const int wave = threadIdx.x >> 5;
  const int half = lane >> 4;
  const int l16  = lane & 15;
  const int h  = blockIdx.y;
  const int qt = blockIdx.x * 8 + wave;
  if (qt >= QT_COUNT) return;
  const int m0 = qt * 16;

  // Q fragments (A, 16x32 each) covering dims 0..31 and 32..63 (scale pre-folded)
  const bf16_t* qrowp = qb + (size_t)min(m0 + l16, S_LEN - 1) * DM + h * HD;
  const v16bf aq0 = load_frag2(qrowp + half * 8,      qrowp + half * 8 + 16);
  const v16bf aq1 = load_frag2(qrowp + 32 + half * 8, qrowp + 32 + half * 8 + 16);

  // per-lane query positions for its 8 C-layout rows (M = r + 8*half)
  int qrow[8], qx[8], qy[8], qz[8];
#pragma unroll
  for (int r = 0; r < 8; ++r) {
    qrow[r] = m0 + r + 8 * half;
    int rr = min(qrow[r], S_LEN - 1);
    qx[r] = pos[rr * 3 + 0]; qy[r] = pos[rr * 3 + 1]; qz[r] = pos[rr * 3 + 2];
  }

  float mi[8], li[8];
  v8f acc[4] = {};
#pragma unroll
  for (int r = 0; r < 8; ++r) { mi[r] = NEGF; li[r] = 0.0f; }

  const int kend = min(m0 + 16, S_LEN);
  const int nchunks = (kend + 31) / 32;
  bf16_t* myp = plds[wave];

  for (int kc = 0; kc < nchunks; ++kc) {
    const int kbase = kc * 32;
    v8f sc[2];
#pragma unroll
    for (int t = 0; t < 2; ++t) {
      const int col = kbase + t * 16 + l16;
      const bf16_t* krow = kbuf + (size_t)min(col, S_LEN - 1) * DM + h * HD + half * 8;
      v16bf bk0 = load_frag2(krow,      krow + 16);       // dims 0..31
      v16bf bk1 = load_frag2(krow + 32, krow + 48);       // dims 32..63
      v8f s = {};
      s = __builtin_amdgcn_wmma_f32_16x16x32_bf16(false, aq0, false, bk0, (short)0, s, false, false);
      s = __builtin_amdgcn_wmma_f32_16x16x32_bf16(false, aq1, false, bk1, (short)0, s, false, false);
      // relative-position bias + causal mask
      const int cc = min(col, S_LEN - 1);
      const int kx = pos[cc * 3 + 0], ky = pos[cc * 3 + 1], kz = pos[cc * 3 + 2];
#pragma unroll
      for (int r = 0; r < 8; ++r) {
        int dx = min(max(qx[r] - kx, -30), 30) + 30;
        int dy = min(max(qy[r] - ky, -30), 30) + 30;
        int dz = min(max(qz[r] - kz,  -8),  8) +  8;
        float b = bias_x[dx * NH + h] + bias_y[dy * NH + h] + bias_z[dz * NH + h];
        float v = s[r] + b;
        if (col > qrow[r] || qrow[r] >= S_LEN || col >= S_LEN) v = NEGF;
        s[r] = v;
      }
      sc[t] = s;
    }
    // online softmax: per-row (half-wave) reductions over the 16 N-lanes
    float alpha[8];
#pragma unroll
    for (int r = 0; r < 8; ++r) {
      float mx = fmaxf(sc[0][r], sc[1][r]);
#pragma unroll
      for (int off = 8; off >= 1; off >>= 1) mx = fmaxf(mx, __shfl_xor(mx, off, 32));
      float mnew = fmaxf(mi[r], mx);
      alpha[r] = __expf(mi[r] - mnew);
      float p0 = __expf(sc[0][r] - mnew);
      float p1 = __expf(sc[1][r] - mnew);
      sc[0][r] = p0; sc[1][r] = p1;
      float rs = p0 + p1;
#pragma unroll
      for (int off = 8; off >= 1; off >>= 1) rs += __shfl_xor(rs, off, 32);
      li[r] = li[r] * alpha[r] + rs;
      mi[r] = mnew;
    }
#pragma unroll
    for (int j = 0; j < 4; ++j)
#pragma unroll
      for (int r = 0; r < 8; ++r) acc[j][r] *= alpha[r];

    // C-layout P -> LDS (row-major 16x32 bf16), re-read as A fragment for PV
#pragma unroll
    for (int r = 0; r < 8; ++r) {
      int row = r + 8 * half;
      myp[row * 32 + l16]      = f2bf(sc[0][r]);
      myp[row * 32 + 16 + l16] = f2bf(sc[1][r]);
    }
    const bf16_t* prow = myp + l16 * 32;
    v16bf ap = load_frag2(prow + half * 8, prow + half * 8 + 16);
#pragma unroll
    for (int j = 0; j < 4; ++j) {
      // B frag from transposed V: contiguous along keys
      const bf16_t* vrow = vt + ((size_t)h * HD + j * 16 + l16) * SP + kbase + half * 8;
      v16bf bv = load_frag2(vrow, vrow + 16);
      acc[j] = __builtin_amdgcn_wmma_f32_16x16x32_bf16(false, ap, false, bv,
                                                       (short)0, acc[j], false, false);
    }
  }
  // normalize and store bf16 attention output [S, H*HD]
#pragma unroll
  for (int r = 0; r < 8; ++r) {
    if (qrow[r] >= S_LEN) continue;
    float rinv = 1.0f / li[r];
#pragma unroll
    for (int j = 0; j < 4; ++j)
      attnb[(size_t)qrow[r] * DM + h * HD + j * 16 + l16] = f2bf(acc[j][r] * rinv);
  }
}

// ------------------------------------------------------------------------------------
extern "C" void kernel_launch(void* const* d_in, const int* in_sizes, int n_in,
                              void* d_out, int out_size, void* d_ws, size_t ws_size,
                              hipStream_t stream) {
  const float* x      = (const float*)d_in[0];
  const float* Wqkv   = (const float*)d_in[1];
  const float* Wout   = (const float*)d_in[2];
  const float* bias_x = (const float*)d_in[3];
  const float* bias_y = (const float*)d_in[4];
  const float* bias_z = (const float*)d_in[5];
  const int*   pos    = (const int*)d_in[6];
  float* out = (float*)d_out;

  char* ws = (char*)d_ws;
  size_t off = 0;
  auto take = [&](size_t bytes) -> void* {
    void* p = ws + off;
    off += (bytes + 255) & ~(size_t)255;
    return p;
  };
  bf16_t* xb    = (bf16_t*)take((size_t)S_LEN * DM * 2);
  bf16_t* wqkvb = (bf16_t*)take((size_t)3 * DM * DM * 2);
  bf16_t* woutb = (bf16_t*)take((size_t)DM * DM * 2);
  float*  qkvf  = (float*) take((size_t)S_LEN * 3 * DM * 4);
  bf16_t* qbuf  = (bf16_t*)take((size_t)S_LEN * DM * 2);
  bf16_t* kbuf  = (bf16_t*)take((size_t)S_LEN * DM * 2);
  bf16_t* vtb   = (bf16_t*)take((size_t)NH * HD * SP * 2);
  bf16_t* attnb = (bf16_t*)take((size_t)S_LEN * DM * 2);

  const int nx = S_LEN * DM, nwq = 3 * DM * DM, nwo = DM * DM;
  k_f32_to_bf16<<<(nx  + 255) / 256, 256, 0, stream>>>(x,    xb,    nx);
  k_f32_to_bf16<<<(nwq + 255) / 256, 256, 0, stream>>>(Wqkv, wqkvb, nwq);
  k_f32_to_bf16<<<(nwo + 255) / 256, 256, 0, stream>>>(Wout, woutb, nwo);

  // qkv = x @ Wqkv^T  : M=1863, N=3072, K=1024
  k_gemm_bf16<<<dim3((S_LEN + 255) / 256, 3 * DM / 64), 256, 0, stream>>>(
      xb, wqkvb, qkvf, S_LEN, 3 * DM, DM);

  k_rope_pack<<<(S_LEN * NH + 255) / 256, 256, 0, stream>>>(qkvf, pos, qbuf, kbuf, vtb);

  k_attention<<<dim3((QT_COUNT + 7) / 8, NH), 256, 0, stream>>>(
      qbuf, kbuf, vtb, bias_x, bias_y, bias_z, pos, attnb);

  // out = attn @ Wout^T : M=1863, N=1024, K=1024
  k_gemm_bf16<<<dim3((S_LEN + 255) / 256, DM / 64), 256, 0, stream>>>(
      attnb, woutb, out, S_LEN, DM, DM);
}
